// SimpleRNNModelTensorFlow_75900662055572
// MI455X (gfx1250) — compile-verified, matrix operand order checked by reading, not run on hardware
//
#include <hip/hip_runtime.h>

// SimpleRNN fused kernel for MI455X (gfx1250, wave32, WMMA).
//
// x: [B=2048, T=256, F=64] f32, Wx: [64,50], Wh: [50,50], b: [50], Wd: [50,1], bd: [1]
// out: [B,1] f32
//
// One block = 16 batch rows, 4 waves (128 threads). Wave w owns hidden
// columns [16w, 16w+16). Hidden dim padded 50->64 with zeros so padded h
// columns stay exactly 0 through relu. Weights live in registers as WMMA B
// fragments for the whole 256-step scan; h carries between steps via LDS f16.

typedef __attribute__((ext_vector_type(16))) _Float16 v16h;
typedef __attribute__((ext_vector_type(8)))  _Float16 v8h;
typedef __attribute__((ext_vector_type(8)))  float    v8f;
typedef __attribute__((ext_vector_type(4)))  float    v4f;

#define HIDDEN 50
#define FEAT   64
#define HPAD   64
#define TSTEPS 256
#define ROWS   16

// Load one 16x32 f16 fragment from a row-major [row][k] f16 array in LDS.
// p must already point at (row_base + k_base + lanehalf*8).
// Layout (ISA 7.12.2, 16-bit A 16x32): lanes 0-15 hold K=0..7 & 16..23,
// lanes 16-31 hold K=8..15 & 24..31, packed 2 halves per VGPR.
__device__ __forceinline__ v16h ld_frag_lds(const _Float16* p) {
    v8h lo = *(const v8h*)(p);        // K = kb + half*8 + 0..7
    v8h hi = *(const v8h*)(p + 16);   // K = kb + half*8 + 16..23
    v16h f;
#pragma unroll
    for (int i = 0; i < 8; ++i) { f[i] = lo[i]; f[8 + i] = hi[i]; }
    return f;
}

// Load one 16x32 A fragment from row-major f32 global memory, cvt to f16.
// p must already point at (row_base + k_base + lanehalf*8).
__device__ __forceinline__ v16h ld_frag_global_f32(const float* p) {
    v4f a0 = *(const v4f*)(p);
    v4f a1 = *(const v4f*)(p + 4);
    v4f a2 = *(const v4f*)(p + 16);
    v4f a3 = *(const v4f*)(p + 20);
    v16h f;
#pragma unroll
    for (int i = 0; i < 4; ++i) {
        f[i]      = (_Float16)a0[i];
        f[4 + i]  = (_Float16)a1[i];
        f[8 + i]  = (_Float16)a2[i];
        f[12 + i] = (_Float16)a3[i];
    }
    return f;
}

extern "C" __global__ __launch_bounds__(128)
void rnn_fused(const float* __restrict__ x,  const float* __restrict__ Wx,
               const float* __restrict__ Wh, const float* __restrict__ b,
               const float* __restrict__ Wd, const float* __restrict__ bd,
               float* __restrict__ out, int B)
{
    __shared__ _Float16 sWxT[HPAD * FEAT];   // WxT[n][k], zero-padded, 8 KB
    __shared__ _Float16 sWhT[HPAD * HPAD];   // WhT[n][k], zero-padded, 8 KB
    __shared__ _Float16 sH[ROWS * HPAD];     // h carry [m][n] f16, 2 KB
    __shared__ float    sB[HPAD];

    const int tid  = threadIdx.x;        // 0..127
    const int wave = tid >> 5;           // 0..3 -> n-tile
    const int lane = tid & 31;
    const int half = lane >> 4;          // lane half selects K sub-block
    const int l16  = lane & 15;
    const size_t batch0 = (size_t)blockIdx.x * ROWS;

    // ---- stage zero-padded, transposed weights into LDS ----
    for (int i = tid; i < HPAD * FEAT; i += 128) {
        int n = i >> 6, k = i & 63;      // sWxT[n][k] = Wx[k][n]
        sWxT[i] = (_Float16)((n < HIDDEN) ? Wx[k * HIDDEN + n] : 0.0f);
    }
    for (int i = tid; i < HPAD * HPAD; i += 128) {
        int n = i >> 6, k = i & 63;      // sWhT[n][k] = Wh[k][n]
        sWhT[i] = (_Float16)((n < HIDDEN && k < HIDDEN) ? Wh[k * HIDDEN + n] : 0.0f);
    }
    if (tid < HPAD) sB[tid] = (tid < HIDDEN) ? b[tid] : 0.0f;
    for (int i = tid; i < ROWS * HPAD; i += 128) sH[i] = (_Float16)0.0f;  // h0 = 0
    __syncthreads();

    // ---- preload B fragments (weights) into registers; reused 256x ----
    const int nb = wave * 16;
    const _Float16* bxRow = sWxT + (size_t)(nb + l16) * FEAT + half * 8;
    const _Float16* bhRow = sWhT + (size_t)(nb + l16) * HPAD + half * 8;
    v16h bx0 = ld_frag_lds(bxRow);        // K = 0..31  of WxT column tile
    v16h bx1 = ld_frag_lds(bxRow + 32);   // K = 32..63
    v16h bh0 = ld_frag_lds(bhRow);        // K = 0..31  of WhT column tile
    v16h bh1 = ld_frag_lds(bhRow + 32);   // K = 32..63
    const float bias = sB[nb + l16];      // per output column N

    const float*    xRow = x + (batch0 + l16) * (size_t)(TSTEPS * FEAT) + half * 8;
    const _Float16* hRow = sH + (size_t)l16 * HPAD + half * 8;

    const int Mbase = half * 8;           // C/D layout: lanes16-31 hold M=r+8
    const int Ncol  = nb + l16;

    // ---- sequential scan over T ----
    for (int t = 0; t < TSTEPS; ++t) {
        const float* xp = xRow + t * FEAT;
        v16h a0 = ld_frag_global_f32(xp);        // x_t  K=0..31
        v16h a1 = ld_frag_global_f32(xp + 32);   // x_t  K=32..63
        v16h h0 = ld_frag_lds(hRow);             // h_{t-1} K=0..31
        v16h h1 = ld_frag_lds(hRow + 32);        // h_{t-1} K=32..63 (pad = 0)

        v8f c = {};
        c = __builtin_amdgcn_wmma_f32_16x16x32_f16(false, a0, false, bx0, (short)0, c, false, false);
        c = __builtin_amdgcn_wmma_f32_16x16x32_f16(false, a1, false, bx1, (short)0, c, false, false);
        c = __builtin_amdgcn_wmma_f32_16x16x32_f16(false, h0, false, bh0, (short)0, c, false, false);
        c = __builtin_amdgcn_wmma_f32_16x16x32_f16(false, h1, false, bh1, (short)0, c, false, false);

        __syncthreads();   // all waves finished reading h_{t-1}
#pragma unroll
        for (int r = 0; r < 8; ++r) {
            float v = c[r] + bias;
            v = v > 0.0f ? v : 0.0f;                   // relu
            sH[(size_t)(Mbase + r) * HPAD + Ncol] = (_Float16)v;
        }
        __syncthreads();   // h_t visible to all waves
    }

    // ---- dense head: out[m] = h_final[m] . Wd + bd ----
    if (tid < ROWS && (batch0 + tid) < (size_t)B) {
        float acc = bd[0];
#pragma unroll 2
        for (int k = 0; k < HIDDEN; ++k)
            acc += (float)sH[(size_t)tid * HPAD + k] * Wd[k];
        out[batch0 + tid] = acc;
    }
}

extern "C" void kernel_launch(void* const* d_in, const int* in_sizes, int n_in,
                              void* d_out, int out_size, void* d_ws, size_t ws_size,
                              hipStream_t stream) {
    const float* x  = (const float*)d_in[0];
    const float* Wx = (const float*)d_in[1];
    const float* Wh = (const float*)d_in[2];
    const float* b  = (const float*)d_in[3];
    const float* Wd = (const float*)d_in[4];
    const float* bd = (const float*)d_in[5];
    float* out = (float*)d_out;
    (void)n_in; (void)out_size; (void)d_ws; (void)ws_size;

    const int B = in_sizes[0] / (TSTEPS * FEAT);   // 2048
    const int grid = (B + ROWS - 1) / ROWS;        // 128 blocks x 128 threads
    hipLaunchKernelGGL(rnn_fused, dim3(grid), dim3(128), 0, stream,
                       x, Wx, Wh, b, Wd, bd, out, B);
}
